// RnnModelInterp_75814762709699
// MI455X (gfx1250) — matrix operand order repaired and checked by reading, not compiled, not run on hardware
//
#include <hip/hip_runtime.h>
#include <math.h>

// RNN scan, persistent kernel: batch rows independent -> each block owns 16
// rows for all 199 steps; state in LDS; f16 transposed weights streamed from
// L2 each step into v_wmma_f32_16x16x32_f16. Heads also via WMMA (Wh 80x512).
// An opaque zero *offset* (not pointer) defeats LICM so weight loads stay in
// the time loop as global_load_b128 (keeping addrspace inference intact).

typedef _Float16 h16;
typedef __attribute__((ext_vector_type(16))) _Float16 v16h;
typedef __attribute__((ext_vector_type(8)))  _Float16 v8h;
typedef __attribute__((ext_vector_type(8)))  float    v8f;

#define T_STEPS 200
#define B_      256
#define H_      512
#define NC_     3
#define NM_     64
#define K0      608   // 96 (padded x: 3 cat + 64 val + 29 zero) + 512 (h0)
#define K1      1024  // 512 (h0_new) + 512 (h1_prev)
#define NH      80    // head cols: 3 (cat) + 64 (val) + 13 zero-pad

// ---- WMMA fragment loaders (row-major f16 source, per CDNA5 ISA layouts) ----

// A-matrix 16x32 f16 (ISA 7.12.2): lanes 0-15 row M=lane, halves 0-7 -> K=k0..k0+7,
// halves 8-15 -> K=k0+16..k0+23; lanes 16-31: +8 on K within each group.
static __device__ __forceinline__ v16h load_fragA(const h16* rowbase, int k0, int lane) {
  int hi = (lane >> 4) & 1;
  const v8h* p0 = (const v8h*)(rowbase + k0 + 8 * hi);
  const v8h* p1 = (const v8h*)(rowbase + k0 + 16 + 8 * hi);
  union { v16h v; v8h h[2]; } u;
  u.h[0] = *p0;
  u.h[1] = *p1;
  return u.v;
}

// B-matrix 32x16 f16: lane%16 = N column; halves 0-7 -> K=k0+8*hi..+7, halves
// 8-15 -> K=k0+16+8*hi..+7 (same K order as A). Source is pre-transposed weight
// Wt[N][K] row-major, so each group of 8 halves is contiguous.
static __device__ __forceinline__ v16h load_fragB(const h16* rowbase, int k0, int lane) {
  int hi = (lane >> 4) & 1;
  const v8h* p0 = (const v8h*)(rowbase + k0 + 8 * hi);
  const v8h* p1 = (const v8h*)(rowbase + k0 + 16 + 8 * hi);
  union { v16h v; v8h h[2]; } u;
  u.h[0] = *p0;
  u.h[1] = *p1;
  return u.v;
}

// ---- one-time prep: transpose + f16-convert all weights ----
__global__ void prep_weights_kernel(const float* __restrict__ W_ih0,
                                    const float* __restrict__ W_hh0,
                                    const float* __restrict__ W_ih1,
                                    const float* __restrict__ W_hh1,
                                    const float* __restrict__ Wc,
                                    const float* __restrict__ Wm,
                                    h16* __restrict__ Wt0, h16* __restrict__ Wt1,
                                    h16* __restrict__ Wh) {
  const int total = H_ * K0 + H_ * K1 + NH * H_;
  for (int idx = blockIdx.x * blockDim.x + threadIdx.x; idx < total;
       idx += gridDim.x * blockDim.x) {
    if (idx < H_ * K0) {
      int n = idx / K0, k = idx % K0;
      float v = 0.f;
      if (k < NC_ + NM_)      v = W_ih0[k * H_ + n];          // cat+val rows
      else if (k >= 96)       v = W_hh0[(k - 96) * H_ + n];   // recurrent
      Wt0[n * K0 + k] = (h16)v;
    } else if (idx < H_ * K0 + H_ * K1) {
      int id2 = idx - H_ * K0;
      int n = id2 / K1, k = id2 % K1;
      float v = (k < H_) ? W_ih1[k * H_ + n] : W_hh1[(k - H_) * H_ + n];
      Wt1[n * K1 + k] = (h16)v;
    } else {
      int id3 = idx - H_ * K0 - H_ * K1;
      int n = id3 / H_, k = id3 % H_;
      float v = 0.f;
      if (n < NC_)            v = Wc[k * NC_ + n];
      else if (n < NC_ + NM_) v = Wm[k * NM_ + (n - NC_)];
      Wh[n * H_ + k] = (h16)v;
    }
  }
}

// ---- persistent scan kernel: 16 blocks x 1024 threads (32 waves) ----
__global__ __launch_bounds__(1024) void rnn_scan_kernel(
    const h16* __restrict__ Wt0, const h16* __restrict__ Wt1,
    const h16* __restrict__ Wh,
    const float* __restrict__ b0, const float* __restrict__ b1,
    const float* __restrict__ bc, const float* __restrict__ bm,
    const float* __restrict__ cat_seq, const float* __restrict__ val_seq,
    float* __restrict__ out) {
  __shared__ h16   S0l[16 * K0];        // [x_pad(96) | h0(512)]         19456 B
  __shared__ h16   S1l[16 * K1];        // [h0_new(512) | h1(512)]       32768 B
  __shared__ float vcur[16 * NM_];      //                                4096 B
  __shared__ float head_out[16 * NH];   // [logits(3) | oval(64) | pad]   5120 B
                                        // total 61440 B

  const int tid  = threadIdx.x;
  const int lane = tid & 31;
  const int wave = tid >> 5;                 // 0..31
  const int m0   = blockIdx.x * 16;
  const int nn   = wave * 16 + (lane & 15);  // output column 0..511
  const int rbase = (lane >> 4) << 3;        // C/D row base within tile

  // ---- init LDS state ----
  for (int idx = tid; idx < 16 * K0; idx += 1024) {
    int r = idx / K0, k = idx % K0, m = m0 + r;
    float v = 0.f;
    if (k < NC_)            v = cat_seq[m * NC_ + k];
    else if (k < NC_ + NM_) v = val_seq[m * NM_ + (k - NC_)];
    S0l[idx] = (h16)v;
  }
  for (int idx = tid; idx < 16 * K1; idx += 1024) S1l[idx] = (h16)0.f;
  for (int idx = tid; idx < 16 * NM_; idx += 1024) {
    int r = idx >> 6, j = idx & (NM_ - 1);
    vcur[idx] = val_seq[(m0 + r) * NM_ + j];
  }
  __syncthreads();

  const float bias0 = b0[nn];
  const float bias1 = b1[nn];
  float hbias = 0.f;
  if (wave < 5) hbias = (nn < NC_) ? bc[nn]
                        : ((nn < NC_ + NM_) ? bm[nn - NC_] : 0.f);
  const h16* arow0 = &S0l[(lane & 15) * K0];
  const h16* arow1 = &S1l[(lane & 15) * K1];
  const h16* arowh = &S1l[(lane & 15) * K1 + H_];

  for (int t = 0; t < T_STEPS - 1; ++t) {
    // Opaque zero offset: defeats LICM (weight loads stay inside the time
    // loop, streamed from L2) without laundering the pointers' address space,
    // so they lower to global_load_b128 (LOADcnt only), not flat_load.
    unsigned zoff = 0;
    asm volatile("" : "+v"(zoff));
    const h16* wp0 = Wt0 + nn * K0 + zoff;
    const h16* wp1 = Wt1 + nn * K1 + zoff;
    const h16* wph = Wh + (wave < 5 ? nn : 0) * H_ + zoff;

    // ---- layer0: h0_new = tanh([x|h0] @ Wt0^T + b0) -> S1l[:,0:512] ----
    v8f c = {};
    for (int k0 = 0; k0 < K0; k0 += 32) {
      v16h a = load_fragA(arow0, k0, lane);
      v16h b = load_fragB(wp0, k0, lane);
      c = __builtin_amdgcn_wmma_f32_16x16x32_f16(false, a, false, b, (short)0, c,
                                                 false, false);
    }
#pragma unroll
    for (int j = 0; j < 8; ++j)
      S1l[(rbase + j) * K1 + nn] = (h16)tanhf(c[j] + bias0);
    __syncthreads();  // b1: h0_new visible

    // ---- layer1: h1_new = tanh([h0_new|h1] @ Wt1^T + b1), acc in regs ----
    v8f d = {};
    for (int k0 = 0; k0 < K1; k0 += 32) {
      v16h a = load_fragA(arow1, k0, lane);
      v16h b = load_fragB(wp1, k0, lane);
      d = __builtin_amdgcn_wmma_f32_16x16x32_f16(false, a, false, b, (short)0, d,
                                                 false, false);
    }
    __syncthreads();  // b2: all waves done reading h1_prev

    // write h1_new into S1l[:,512:]; carry h0_new -> S0l[:,96:608]
#pragma unroll
    for (int j = 0; j < 8; ++j)
      S1l[(rbase + j) * K1 + H_ + nn] = (h16)tanhf(d[j] + bias1);
    for (int idx = tid; idx < 16 * H_; idx += 1024) {
      int r = idx >> 9, k = idx & (H_ - 1);
      S0l[r * K0 + 96 + k] = S1l[r * K1 + k];
    }
    __syncthreads();  // b3: h1_new visible

    // ---- heads via WMMA: [16x512] @ [512x80] -> head_out (+bias fused) ----
    if (wave < 5) {
      v8f e = {};
      for (int k0 = 0; k0 < H_; k0 += 32) {
        v16h a = load_fragA(arowh, k0, lane);
        v16h b = load_fragB(wph, k0, lane);
        e = __builtin_amdgcn_wmma_f32_16x16x32_f16(false, a, false, b, (short)0, e,
                                                   false, false);
      }
#pragma unroll
      for (int j = 0; j < 8; ++j)
        head_out[(rbase + j) * NH + nn] = e[j] + hbias;
    }
    __syncthreads();  // b4: head_out visible

    // ---- softmax + output + cat imputation (one thread per row) ----
    if (tid < 16) {
      int r = tid, m = m0 + r;
      float l0 = head_out[r * NH + 0], l1 = head_out[r * NH + 1],
            l2 = head_out[r * NH + 2];
      float mx = fmaxf(l0, fmaxf(l1, l2));
      float e0 = __expf(l0 - mx), e1 = __expf(l1 - mx), e2 = __expf(l2 - mx);
      float inv = 1.f / (e0 + e1 + e2);
      float o0 = e0 * inv, o1 = e1 * inv, o2 = e2 * inv;
      float* orow = out + (size_t)t * B_ * NC_ + m * NC_;
      orow[0] = o0; orow[1] = o1; orow[2] = o2;
      const float* cn = cat_seq + (size_t)(t + 1) * B_ * NC_ + m * NC_;
      float c0 = cn[0], c1 = cn[1], c2v = cn[2];
      if (__builtin_isnan(c0)) c0 = o0;
      if (__builtin_isnan(c1)) c1 = o1;
      if (__builtin_isnan(c2v)) c2v = o2;
      S0l[r * K0 + 0] = (h16)c0;
      S0l[r * K0 + 1] = (h16)c1;
      S0l[r * K0 + 2] = (h16)c2v;
    }
    // ---- val imputation: o_val = head + vcur; next x-val cols + f32 carry ----
    for (int idx = tid; idx < 16 * NM_; idx += 1024) {
      int r = idx >> 6, j = idx & (NM_ - 1), m = m0 + r;
      float ov = head_out[r * NH + NC_ + j] + vcur[idx];
      float vn = val_seq[(size_t)(t + 1) * B_ * NM_ + m * NM_ + j];
      if (__builtin_isnan(vn)) vn = ov;
      vcur[idx] = vn;
      S0l[r * K0 + NC_ + j] = (h16)vn;
    }
    __syncthreads();  // b5: S0l ready for next step
  }
}

extern "C" void kernel_launch(void* const* d_in, const int* in_sizes, int n_in,
                              void* d_out, int out_size, void* d_ws, size_t ws_size,
                              hipStream_t stream) {
  const float* cat_seq = (const float*)d_in[0];
  const float* val_seq = (const float*)d_in[1];
  const float* W_ih0   = (const float*)d_in[2];
  const float* W_hh0   = (const float*)d_in[3];
  const float* b0      = (const float*)d_in[4];
  const float* W_ih1   = (const float*)d_in[5];
  const float* W_hh1   = (const float*)d_in[6];
  const float* b1      = (const float*)d_in[7];
  const float* Wc      = (const float*)d_in[8];
  const float* bc      = (const float*)d_in[9];
  const float* Wm      = (const float*)d_in[10];
  const float* bm      = (const float*)d_in[11];
  float* out = (float*)d_out;

  // workspace: transposed f16 weights
  char* ws = (char*)d_ws;
  h16* Wt0 = (h16*)(ws + 0);         // 512*608*2  = 622592 B
  h16* Wt1 = (h16*)(ws + 622592);    // 512*1024*2 = 1048576 B
  h16* Wh  = (h16*)(ws + 1671168);   // 80*512*2   = 81920 B  -> total ~1.75MB

  prep_weights_kernel<<<512, 256, 0, stream>>>(W_ih0, W_hh0, W_ih1, W_hh1,
                                               Wc, Wm, Wt0, Wt1, Wh);
  rnn_scan_kernel<<<16, 1024, 0, stream>>>(Wt0, Wt1, Wh, b0, b1, bc, bm,
                                           cat_seq, val_seq, out);
}
